// FieldAttentionModule_32014686224409
// MI455X (gfx1250) — matrix operand order repaired and checked by reading, not compiled, not run on hardware
//
#include <hip/hip_runtime.h>
#include <hip/hip_bf16.h>

typedef _Float16 half8_t  __attribute__((ext_vector_type(8)));
typedef _Float16 half16_t __attribute__((ext_vector_type(16)));
typedef float    floatx8  __attribute__((ext_vector_type(8)));

union V16 { half16_t v; half8_t h[2]; };

static __device__ __forceinline__ floatx8 wmma_f16(half16_t a, half16_t b, floatx8 c) {
    return __builtin_amdgcn_wmma_f32_16x16x32_f16(false, a, false, b, (short)0, c, false, false);
}
static __device__ __forceinline__ void lds_fence() {
    asm volatile("s_wait_dscnt 0" ::: "memory");
}
static __device__ __forceinline__ void async_fence() {
    asm volatile("s_wait_asynccnt 0" ::: "memory");
}
// Async DMA: 16 bytes global -> LDS, per lane. VDST = LDS byte address (VGPR),
// VADDR = 64-bit global address (GV mode, no SADDR). Tracked by ASYNCcnt.
static __device__ __forceinline__ void async_g2l_b128(unsigned lds_off, const void* gaddr) {
    asm volatile("global_load_async_to_lds_b128 %0, %1, off"
                 :: "v"(lds_off), "v"(gaddr) : "memory");
}

// Problem constants
#define NTOK   65536   // V*T*N*TT*NN*DD tokens
#define NWIN   512     // T*N windows

// token linear index (window-major): ((((T*256+n)*2+v)*4+t)*4+nn)*4+d
// x element index: ((((v*8 + T*4+t)*1024 + n*4+nn)*4 + d)*64 + f

// ---------------------------------------------------------------------------
// Kernel A: LN1 + Q/K/V projection (f32 LN, f16 WMMA, f32 accum)
// 16 tokens per block, 128 threads (4 waves)
// ---------------------------------------------------------------------------
__global__ void __launch_bounds__(128)
k_ln_qkv(const float* __restrict__ x, const float* __restrict__ ln_s,
         const float* __restrict__ ln_b, const float* __restrict__ Wq,
         const float* __restrict__ Wkv, const float* __restrict__ bkv,
         _Float16* __restrict__ qws, _Float16* __restrict__ kws,
         _Float16* __restrict__ vws)
{
    __shared__ _Float16 hbuf[16 * 64];     // LN'd activations, f16
    __shared__ _Float16 WqT[128 * 64];     // [n][k] transposed weights
    __shared__ _Float16 WkvT[256 * 64];

    int tid = threadIdx.x;
    for (int i = tid; i < 128 * 64; i += 128) {
        int nn_ = i >> 6, k = i & 63;
        WqT[i] = (_Float16)Wq[k * 128 + nn_];
    }
    for (int i = tid; i < 256 * 64; i += 128) {
        int nn_ = i >> 6, k = i & 63;
        WkvT[i] = (_Float16)Wkv[k * 256 + nn_];
    }

    // ---- LayerNorm: 16 tokens x (8 threads x 8 feats) ----
    int tok_local = tid >> 3;
    int fg = (tid & 7) * 8;
    int tok = blockIdx.x * 16 + tok_local;
    int d  = tok & 3, nn = (tok >> 2) & 3, t = (tok >> 4) & 3;
    int v  = (tok >> 6) & 1, n = (tok >> 7) & 255, Tt = (tok >> 15) & 1;
    long xrow = ((((long)v * 8 + (Tt * 4 + t)) * 1024 + (n * 4 + nn)) * 4 + d) * 64;

    float vals[8], s = 0.f, s2 = 0.f;
    for (int j = 0; j < 8; ++j) {
        float xv = x[xrow + fg + j];
        vals[j] = xv; s += xv; s2 += xv * xv;
    }
    for (int m = 1; m < 8; m <<= 1) { s += __shfl_xor(s, m, 32); s2 += __shfl_xor(s2, m, 32); }
    float mean = s * (1.f / 64.f);
    float var  = s2 * (1.f / 64.f) - mean * mean;
    float rstd = rsqrtf(var + 1e-5f);
    for (int j = 0; j < 8; ++j) {
        int f = fg + j;
        hbuf[tok_local * 64 + f] = (_Float16)((vals[j] - mean) * rstd * ln_s[f] + ln_b[f]);
    }
    __syncthreads();

    // ---- WMMA projections ----
    int wave = tid >> 5, lane = tid & 31;
    int row = lane & 15, hi = lane >> 4;

    V16 a0, a1;   // A tiles for K=[0,32) and [32,64)
    a0.h[0] = *(const half8_t*)&hbuf[row * 64 +  0 + hi * 8];
    a0.h[1] = *(const half8_t*)&hbuf[row * 64 + 16 + hi * 8];
    a1.h[0] = *(const half8_t*)&hbuf[row * 64 + 32 + hi * 8];
    a1.h[1] = *(const half8_t*)&hbuf[row * 64 + 48 + hi * 8];

    long tokbase = (long)blockIdx.x * 16;

    // q: 8 col tiles, 2 per wave
    for (int it = 0; it < 2; ++it) {
        int nb = (wave * 2 + it) * 16;
        V16 b0, b1;
        const _Float16* wt = &WqT[(nb + row) * 64];
        b0.h[0] = *(const half8_t*)&wt[hi * 8];      b0.h[1] = *(const half8_t*)&wt[16 + hi * 8];
        b1.h[0] = *(const half8_t*)&wt[32 + hi * 8]; b1.h[1] = *(const half8_t*)&wt[48 + hi * 8];
        floatx8 c = {};
        c = wmma_f16(a0.v, b0.v, c);
        c = wmma_f16(a1.v, b1.v, c);
        for (int r = 0; r < 8; ++r)
            qws[(tokbase + r + 8 * hi) * 128 + nb + row] = (_Float16)c[r];
    }
    // kv: 16 col tiles, 4 per wave (bias in accumulator init)
    for (int it = 0; it < 4; ++it) {
        int nb = (wave * 4 + it) * 16;
        V16 b0, b1;
        const _Float16* wt = &WkvT[(nb + row) * 64];
        b0.h[0] = *(const half8_t*)&wt[hi * 8];      b0.h[1] = *(const half8_t*)&wt[16 + hi * 8];
        b1.h[0] = *(const half8_t*)&wt[32 + hi * 8]; b1.h[1] = *(const half8_t*)&wt[48 + hi * 8];
        floatx8 c;
        float bb = bkv[nb + row];
        for (int r = 0; r < 8; ++r) c[r] = bb;
        c = wmma_f16(a0.v, b0.v, c);
        c = wmma_f16(a1.v, b1.v, c);
        int col = nb + row;
        for (int r = 0; r < 8; ++r) {
            long tgt = (tokbase + r + 8 * hi) * 128;
            if (col < 128) kws[tgt + col]       = (_Float16)c[r];
            else           vws[tgt + col - 128] = (_Float16)c[r];
        }
    }
}

// ---------------------------------------------------------------------------
// Halo mapping: window (Tw,n), kv row r in [0,512) -> source token index.
// Returns false for halo-zero rows (which collapse to the bkv bias row).
// ---------------------------------------------------------------------------
static __device__ __forceinline__ bool kv_src(int Tw, int n, int r, long& src) {
    int delta = r & 7, nnq = (r >> 3) & 3, tau = (r >> 5) & 7, vv = r >> 8;
    int Tsrc, tsrc;
    if (tau < 2)      { Tsrc = Tw - 1; tsrc = tau + 2; }
    else if (tau < 6) { Tsrc = Tw;     tsrc = tau - 2; }
    else              { Tsrc = Tw + 1; tsrc = tau - 6; }
    if (delta < 2 || delta >= 6 || Tsrc < 0 || Tsrc > 1) return false;
    src = ((((long)(Tsrc * 256 + n) * 2 + vv) * 4 + tsrc) * 4 + nnq) * 4 + (delta - 2);
    return true;
}

// ---------------------------------------------------------------------------
// Kernel B: windowed flash attention. 1 block / window, 1 wave / head.
// Per pass: 2 q-tiles (32 rows) resident, stream kv in chunks of 32.
// K chunk staged via async global->LDS DMA (ASYNCcnt), V staged transposed.
// ---------------------------------------------------------------------------
__global__ void __launch_bounds__(128)
k_attn(const _Float16* __restrict__ qws, const _Float16* __restrict__ kws,
       const _Float16* __restrict__ vws, const float* __restrict__ bkv,
       _Float16* __restrict__ ows)
{
    __shared__ _Float16 Kbuf[4][32 * 32];   // per-wave K chunk [kv][H]
    __shared__ _Float16 Vtb [4][32 * 40];   // per-wave V chunk transposed [H][kv] (pad 40)
    __shared__ _Float16 Pbuf[4][32 * 32];   // per-wave P (2 q-tiles x 32 kv)

    int w = blockIdx.x;
    int Tw = w >> 8, n = w & 255;
    int wave = threadIdx.x >> 5, lane = threadIdx.x & 31;
    int h = wave;                     // head
    int row = lane & 15, hi = lane >> 4;
    const float scale = 0.17677669529663687f;   // 1/sqrt(32)

    _Float16* Kb = Kbuf[wave];
    _Float16* Vb = Vtb[wave];
    _Float16* Pb = Pbuf[wave];
    unsigned kb_lds = (unsigned)(uintptr_t)&Kb[lane * 32];   // LDS byte addr of lane's K row
    long wq = (long)w * 128;

    for (int mp = 0; mp < 4; ++mp) {
        V16 aq[2];
        for (int u = 0; u < 2; ++u) {
            const _Float16* qp = &qws[(wq + (mp * 2 + u) * 16 + row) * 128 + h * 32];
            aq[u].h[0] = *(const half8_t*)&qp[hi * 8];
            aq[u].h[1] = *(const half8_t*)&qp[16 + hi * 8];
        }
        if (mp < 3)   // prefetch next pass' q rows (global_prefetch_b8)
            __builtin_prefetch(&qws[(wq + (mp * 2 + 2) * 16 + row) * 128 + h * 32], 0, 3);

        floatx8 acc[2][2] = {};
        float mr[2][8], lr[2][8];
        for (int u = 0; u < 2; ++u)
            for (int r = 0; r < 8; ++r) { mr[u][r] = -1e30f; lr[u][r] = 0.f; }

        for (int ch = 0; ch < 16; ++ch) {
            // ---- stage kv chunk: lane = chunk-local kv row ----
            long src;
            if (kv_src(Tw, n, ch * 32 + lane, src)) {
                const _Float16* kp = &kws[src * 128 + h * 32];
                const _Float16* vp = &vws[src * 128 + h * 32];
                // K row: 64B async DMA straight into LDS, no VGPR round trip
                for (int j = 0; j < 4; ++j)
                    async_g2l_b128(kb_lds + j * 16, kp + j * 8);
                // V row: load then scatter transposed ([H][kv]) for the B operand
                _Float16 vrow[32];
                for (int j = 0; j < 4; ++j)
                    *(half8_t*)&vrow[j * 8] = *(const half8_t*)&vp[j * 8];
                for (int j = 0; j < 32; ++j)
                    Vb[j * 40 + lane] = vrow[j];
            } else {   // halo-zero token -> projection collapses to bias row
                for (int j = 0; j < 32; ++j) {
                    Kb[lane * 32 + j] = (_Float16)bkv[h * 32 + j];
                    Vb[j * 40 + lane] = (_Float16)bkv[128 + h * 32 + j];
                }
            }
            // prefetch next chunk's kv source rows into cache
            long nsrc;
            if (ch < 15 && kv_src(Tw, n, (ch + 1) * 32 + lane, nsrc)) {
                __builtin_prefetch(&kws[nsrc * 128 + h * 32], 0, 3);
                __builtin_prefetch(&vws[nsrc * 128 + h * 32], 0, 3);
            }
            async_fence();   // s_wait_asynccnt 0 : async K DMA done
            lds_fence();     // s_wait_dscnt 0   : V/bias ds_stores done

            // ---- scores + online softmax ----
            for (int u = 0; u < 2; ++u) {
                V16 bk0, bk1;
                const _Float16* kp0 = &Kb[row * 32];
                const _Float16* kp1 = &Kb[(16 + row) * 32];
                bk0.h[0] = *(const half8_t*)&kp0[hi * 8]; bk0.h[1] = *(const half8_t*)&kp0[16 + hi * 8];
                bk1.h[0] = *(const half8_t*)&kp1[hi * 8]; bk1.h[1] = *(const half8_t*)&kp1[16 + hi * 8];
                floatx8 s0 = {}, s1 = {};
                s0 = wmma_f16(aq[u].v, bk0.v, s0);
                s1 = wmma_f16(aq[u].v, bk1.v, s1);
                for (int rr = 0; rr < 8; ++rr) {
                    float x0 = s0[rr] * scale, x1 = s1[rr] * scale;
                    float mx = fmaxf(x0, x1);
                    for (int mk = 1; mk < 16; mk <<= 1) mx = fmaxf(mx, __shfl_xor(mx, mk, 32));
                    float mnew  = fmaxf(mr[u][rr], mx);
                    float alpha = __expf(mr[u][rr] - mnew);
                    float p0 = __expf(x0 - mnew);
                    float p1 = __expf(x1 - mnew);
                    float rs = p0 + p1;
                    for (int mk = 1; mk < 16; mk <<= 1) rs += __shfl_xor(rs, mk, 32);
                    lr[u][rr] = lr[u][rr] * alpha + rs;
                    mr[u][rr] = mnew;
                    acc[u][0][rr] *= alpha;
                    acc[u][1][rr] *= alpha;
                    int prow = u * 16 + rr + 8 * hi;
                    Pb[prow * 32 + row]      = (_Float16)p0;
                    Pb[prow * 32 + 16 + row] = (_Float16)p1;
                }
            }
            lds_fence();

            // ---- acc += P @ V ----
            for (int u = 0; u < 2; ++u) {
                V16 ap, bv0, bv1;
                const _Float16* pp = &Pb[(u * 16 + row) * 32];
                ap.h[0] = *(const half8_t*)&pp[hi * 8];
                ap.h[1] = *(const half8_t*)&pp[16 + hi * 8];
                const _Float16* vp0 = &Vb[row * 40];
                const _Float16* vp1 = &Vb[(16 + row) * 40];
                bv0.h[0] = *(const half8_t*)&vp0[hi * 8]; bv0.h[1] = *(const half8_t*)&vp0[16 + hi * 8];
                bv1.h[0] = *(const half8_t*)&vp1[hi * 8]; bv1.h[1] = *(const half8_t*)&vp1[16 + hi * 8];
                acc[u][0] = wmma_f16(ap.v, bv0.v, acc[u][0]);
                acc[u][1] = wmma_f16(ap.v, bv1.v, acc[u][1]);
            }
        }

        // ---- finalize ----
        for (int u = 0; u < 2; ++u) {
            for (int rr = 0; rr < 8; ++rr) {
                float inv = 1.f / lr[u][rr];
                int srow = (mp * 2 + u) * 16 + rr + 8 * hi;
                long ob = (wq + srow) * 128 + h * 32;
                ows[ob + row]      = (_Float16)(acc[u][0][rr] * inv);
                ows[ob + 16 + row] = (_Float16)(acc[u][1][rr] * inv);
            }
        }
    }
}

// ---------------------------------------------------------------------------
// Kernel C: Wo proj + residual + LN2 + MLP(GELU) + residual + output permute
// 16 tokens per block, 128 threads (4 waves)
// ---------------------------------------------------------------------------
__global__ void __launch_bounds__(128)
k_out_mlp(const float* __restrict__ x, const _Float16* __restrict__ ows,
          const float* __restrict__ Wo, const float* __restrict__ bo,
          const float* __restrict__ gamma, const float* __restrict__ ln2_s,
          const float* __restrict__ ln2_b, const float* __restrict__ W1,
          const float* __restrict__ b1, const float* __restrict__ W2,
          const float* __restrict__ b2, const float* __restrict__ gamma_mlp,
          float* __restrict__ out)
{
    __shared__ _Float16 WoT[64 * 128];     // [f][a]
    __shared__ _Float16 W1T[128 * 64];     // [a][f]
    __shared__ _Float16 W2T[64 * 128];     // [f][a]
    __shared__ float    tokbuf[16 * 64];
    __shared__ _Float16 h2buf[16 * 64];
    __shared__ _Float16 mbuf[16 * 128];

    int tid = threadIdx.x;
    for (int i = tid; i < 64 * 128; i += 128) { int nn_ = i >> 7, k = i & 127; WoT[i] = (_Float16)Wo[k * 64 + nn_]; }
    for (int i = tid; i < 128 * 64; i += 128) { int nn_ = i >> 6, k = i & 63;  W1T[i] = (_Float16)W1[k * 128 + nn_]; }
    for (int i = tid; i < 64 * 128; i += 128) { int nn_ = i >> 7, k = i & 127; W2T[i] = (_Float16)W2[k * 64 + nn_]; }
    __syncthreads();

    int wave = tid >> 5, lane = tid & 31;
    int row = lane & 15, hi = lane >> 4;
    long tb = (long)blockIdx.x * 16;
    int nb = wave * 16;   // this wave's 16 output features

    // ---- upd = o @ Wo + bo ----
    floatx8 c;
    float bias = bo[nb + row];
    for (int r = 0; r < 8; ++r) c[r] = bias;
    for (int kb = 0; kb < 128; kb += 32) {
        V16 a, b;
        const _Float16* op = &ows[(tb + row) * 128 + kb];
        a.h[0] = *(const half8_t*)&op[hi * 8];
        a.h[1] = *(const half8_t*)&op[16 + hi * 8];
        const _Float16* wp = &WoT[(nb + row) * 128 + kb];
        b.h[0] = *(const half8_t*)&wp[hi * 8];
        b.h[1] = *(const half8_t*)&wp[16 + hi * 8];
        c = wmma_f16(a.v, b.v, c);
    }
    // ---- residual into tokbuf ----
    for (int r = 0; r < 8; ++r) {
        int rl = r + 8 * hi;
        int tok = (int)(tb + rl);
        int dd = tok & 3, nn2 = (tok >> 2) & 3, tq = (tok >> 4) & 3;
        int vv = (tok >> 6) & 1, nq = (tok >> 7) & 255, Tq = (tok >> 15) & 1;
        long xrow = ((((long)vv * 8 + (Tq * 4 + tq)) * 1024 + (nq * 4 + nn2)) * 4 + dd) * 64;
        int f = nb + row;
        tokbuf[rl * 64 + f] = x[xrow + f] + gamma[f] * c[r];
    }
    __syncthreads();

    // ---- LN2 ----
    int tok_local = tid >> 3;
    int fg = (tid & 7) * 8;
    float vals[8], s = 0.f, s2 = 0.f;
    for (int j = 0; j < 8; ++j) {
        float xv = tokbuf[tok_local * 64 + fg + j];
        vals[j] = xv; s += xv; s2 += xv * xv;
    }
    for (int m = 1; m < 8; m <<= 1) { s += __shfl_xor(s, m, 32); s2 += __shfl_xor(s2, m, 32); }
    float mean = s * (1.f / 64.f);
    float var  = s2 * (1.f / 64.f) - mean * mean;
    float rstd = rsqrtf(var + 1e-5f);
    for (int j = 0; j < 8; ++j) {
        int f = fg + j;
        h2buf[tok_local * 64 + f] = (_Float16)((vals[j] - mean) * rstd * ln2_s[f] + ln2_b[f]);
    }
    __syncthreads();

    // ---- mlp1 + GELU(tanh approx, matches jax default) ----
    V16 a0, a1;
    a0.h[0] = *(const half8_t*)&h2buf[row * 64 +  0 + hi * 8];
    a0.h[1] = *(const half8_t*)&h2buf[row * 64 + 16 + hi * 8];
    a1.h[0] = *(const half8_t*)&h2buf[row * 64 + 32 + hi * 8];
    a1.h[1] = *(const half8_t*)&h2buf[row * 64 + 48 + hi * 8];
    for (int it = 0; it < 2; ++it) {
        int nb1 = (wave * 2 + it) * 16;
        floatx8 cc;
        float bb = b1[nb1 + row];
        for (int r = 0; r < 8; ++r) cc[r] = bb;
        V16 bb0, bb1;
        const _Float16* wp = &W1T[(nb1 + row) * 64];
        bb0.h[0] = *(const half8_t*)&wp[hi * 8];      bb0.h[1] = *(const half8_t*)&wp[16 + hi * 8];
        bb1.h[0] = *(const half8_t*)&wp[32 + hi * 8]; bb1.h[1] = *(const half8_t*)&wp[48 + hi * 8];
        cc = wmma_f16(a0.v, bb0.v, cc);
        cc = wmma_f16(a1.v, bb1.v, cc);
        for (int r = 0; r < 8; ++r) {
            float u_ = cc[r];
            float g = 0.5f * u_ * (1.f + tanhf(0.7978845608028654f * (u_ + 0.044715f * u_ * u_ * u_)));
            mbuf[(r + 8 * hi) * 128 + nb1 + row] = (_Float16)g;
        }
    }
    __syncthreads();

    // ---- mlp2 ----
    floatx8 c2;
    float bb2 = b2[nb + row];
    for (int r = 0; r < 8; ++r) c2[r] = bb2;
    for (int kb = 0; kb < 128; kb += 32) {
        V16 a, b;
        const _Float16* mp = &mbuf[row * 128 + kb];
        a.h[0] = *(const half8_t*)&mp[hi * 8];
        a.h[1] = *(const half8_t*)&mp[16 + hi * 8];
        const _Float16* wp = &W2T[(nb + row) * 128 + kb];
        b.h[0] = *(const half8_t*)&wp[hi * 8];
        b.h[1] = *(const half8_t*)&wp[16 + hi * 8];
        c2 = wmma_f16(a.v, b.v, c2);
    }
    // ---- final residual + permuted store ----
    for (int r = 0; r < 8; ++r) {
        int rl = r + 8 * hi;
        int tok = (int)(tb + rl);
        int dd = tok & 3, nn2 = (tok >> 2) & 3, tq = (tok >> 4) & 3;
        int vv = (tok >> 6) & 1, nq = (tok >> 7) & 255, Tq = (tok >> 15) & 1;
        long xrow = ((((long)vv * 8 + (Tq * 4 + tq)) * 1024 + (nq * 4 + nn2)) * 4 + dd) * 64;
        int f = nb + row;
        out[xrow + f] = tokbuf[rl * 64 + f] + gamma_mlp[f] * c2[r];
    }
}

// ---------------------------------------------------------------------------
extern "C" void kernel_launch(void* const* d_in, const int* in_sizes, int n_in,
                              void* d_out, int out_size, void* d_ws, size_t ws_size,
                              hipStream_t stream) {
    const float* x        = (const float*)d_in[0];
    const float* ln1_s    = (const float*)d_in[1];
    const float* ln1_b    = (const float*)d_in[2];
    const float* Wq       = (const float*)d_in[3];
    const float* Wkv      = (const float*)d_in[4];
    const float* bkv      = (const float*)d_in[5];
    const float* Wo       = (const float*)d_in[6];
    const float* bo       = (const float*)d_in[7];
    const float* gamma    = (const float*)d_in[8];
    const float* ln2_s    = (const float*)d_in[9];
    const float* ln2_b    = (const float*)d_in[10];
    const float* W1       = (const float*)d_in[11];
    const float* b1       = (const float*)d_in[12];
    const float* W2       = (const float*)d_in[13];
    const float* b2       = (const float*)d_in[14];
    const float* gamma_m  = (const float*)d_in[15];
    float* out = (float*)d_out;

    _Float16* qws = (_Float16*)d_ws;
    _Float16* kws = qws + (size_t)NTOK * 128;
    _Float16* vws = kws + (size_t)NTOK * 128;
    _Float16* ows = vws + (size_t)NTOK * 128;

    k_ln_qkv<<<NTOK / 16, 128, 0, stream>>>(x, ln1_s, ln1_b, Wq, Wkv, bkv, qws, kws, vws);
    k_attn<<<NWIN, 128, 0, stream>>>(qws, kws, vws, bkv, ows);
    k_out_mlp<<<NTOK / 16, 128, 0, stream>>>(x, ows, Wo, bo, gamma, ln2_s, ln2_b,
                                             W1, b1, W2, b2, gamma_m, out);
}
